// DCN_layer_5703716569338
// MI455X (gfx1250) — matrix operand order repaired and checked by reading, not compiled, not run on hardware
//
#include <hip/hip_runtime.h>

// ---------------- problem constants (from reference setup_inputs) -----------
#define B_    4
#define C_    64      // input channels of deform conv
#define O_    128     // output channels of deform conv
#define H_    128
#define W_    128
#define HW_   16384   // H_*W_
#define K_    9       // 3x3 taps
#define KC_   576     // C_*K_  (deform-conv GEMM K dimension)
#define OMCH  27      // offset/mask conv output channels (3*K)
#define OMM   32      // OMCH padded to WMMA M granularity
#define OMK   1152    // 2*C_*9 (offset-conv GEMM K dimension)
#define OMKCH 96      // K-chunk staged in LDS for stage-1 WMMA
#define OMROW 98      // padded LDS row stride (even -> 8B-aligned float2)

// ---------------- CDNA5 WMMA vector types ----------------------------------
typedef __attribute__((ext_vector_type(16))) __bf16 v16bf_t;
typedef __attribute__((ext_vector_type(8)))  __bf16 v8bf_t;
typedef __attribute__((ext_vector_type(8)))  float  f32x8_t;
typedef __attribute__((ext_vector_type(2)))  float  f32x2_t;

static __device__ __forceinline__ __bf16 f2bf(float x) {
  // round-to-nearest-even f32 -> bf16 via integer math
  unsigned u = __builtin_bit_cast(unsigned, x);
  u += 0x7FFFu + ((u >> 16) & 1u);
  unsigned short s = (unsigned short)(u >> 16);
  return __builtin_bit_cast(__bf16, s);
}

// ============================================================================
// Stage 0a: pack deform-conv weight [O][C][3][3] -> bf16 [O][KC_] row-major
// ============================================================================
__global__ void pack_w_kernel(const float* __restrict__ w,
                              __bf16* __restrict__ wb, int n) {
  int i = blockIdx.x * blockDim.x + threadIdx.x;
  if (i < n) wb[i] = f2bf(w[i]);
}

// ============================================================================
// Stage 0b: pack offset/mask conv weight [27][1152] -> f32 [32][1152],
// rows 27..31 zero (WMMA M padding). Same linear (co-major) layout.
// ============================================================================
__global__ void pack_wom_kernel(const float* __restrict__ w,
                                float* __restrict__ wpk) {
  int i = blockIdx.x * blockDim.x + threadIdx.x;      // 0 .. 32*1152-1
  if (i < OMM * OMK) wpk[i] = (i < OMCH * OMK) ? w[i] : 0.0f;
}

// ============================================================================
// Stage 1: offset/mask conv as implicit GEMM on V_WMMA_F32_16X16X4_F32.
// Full f32 precision (offsets feed bilinear sampling -> accuracy matters)
// at matrix-pipe throughput (~32x the FLOP/instr of VALU f32).
//   Block = 64 pixels (one contiguous x-run). 8 waves = 2 Mtiles x 4 Ntiles.
//   K = 1152 processed in 12 chunks of 96; im2col slab staged in LDS.
// ============================================================================
__global__ __launch_bounds__(256) void om_wmma_kernel(
    const float* __restrict__ xin, const float* __restrict__ inter,
    const float* __restrict__ wpk, const float* __restrict__ b_om,
    float* __restrict__ om) {
  __shared__ __align__(16) float bs[64 * OMROW];      // 64 px x 96 k (padded)

  const int tid   = threadIdx.x;
  const int pbase = blockIdx.x << 6;                  // 64 pixels per block
  const int b     = pbase >> 14;
  const int rem0  = pbase & (HW_ - 1);
  const int y     = rem0 >> 7;                        // uniform row
  const int x0    = rem0 & (W_ - 1);

  const int wave  = tid >> 5;
  const int lane  = tid & 31;
  const int mtile = wave >> 2;                        // 0..1 (M rows 0-15/16-31)
  const int ntile = wave & 3;                         // 0..3 (16-pixel group)
  const int mrow  = (mtile << 4) + (lane & 15);       // A row (out channel)
  const int half  = lane >> 4;                        // 0: K0,K1  1: K2,K3
  const int pixl  = (ntile << 4) + (lane & 15);       // B column (pixel)

  f32x8_t acc = {};
  for (int cb = 0; cb < OMK; cb += OMKCH) {           // 12 K-chunks
    // ---- fill im2col slab: bs[pix][lk] = feat[b][ci][y+dy][x+dx] ----
    for (int i = 0; i < (64 * OMKCH) / 256; ++i) {    // 24 entries per thread
      int q   = tid + (i << 8);
      int pix = q & 63;
      int lk  = q >> 6;                               // 0..95
      int k2  = cb + lk;                              // ci*9 + t
      int ci  = k2 / 9, t = k2 - ci * 9;
      const float* src = (ci < C_)
          ? (xin   + ((size_t)(b * C_ + ci)      << 14))
          : (inter + ((size_t)(b * C_ + ci - C_) << 14));
      int yy = y + t / 3 - 1;
      int xx = x0 + pix + t % 3 - 1;
      bool ok = ((unsigned)yy < (unsigned)H_) & ((unsigned)xx < (unsigned)W_);
      bs[pix * OMROW + lk] = ok ? src[yy * W_ + xx] : 0.0f;
    }
    __syncthreads();

    // ---- 24 x V_WMMA_F32_16X16X4_F32 over this chunk ----
    const float* arow = wpk + (size_t)mrow * OMK + cb + (half << 1);
    const float* brow = bs + pixl * OMROW + (half << 1);
#pragma unroll
    for (int kkl = 0; kkl < OMKCH / 4; ++kkl) {
      // A 16x4 f32 layout: lanes0-15 hold (K0,K1), lanes16-31 hold (K2,K3)
      f32x2_t a = *(const f32x2_t*)(arow + (kkl << 2));
      // B 4x16 f32 (dual layout): lane = column, halves as above
      f32x2_t bf = *(const f32x2_t*)(brow + (kkl << 2));
      acc = __builtin_amdgcn_wmma_f32_16x16x4_f32(
          /*neg_a=*/false, a, /*neg_b=*/false, bf,
          /*c_mod=*/(short)0, acc, /*reuse_a=*/false, /*reuse_b=*/false);
    }
    __syncthreads();
  }

  // ---- writeout: bias + sigmoid(mask third); rows >= 27 are padding ----
  const int rem = rem0 + pixl;
#pragma unroll
  for (int r = 0; r < 8; ++r) {
    int m = (mtile << 4) + (half << 3) + r;           // f32 C/D layout
    if (m < OMCH) {
      float a = acc[r] + b_om[m];
      if (m >= 18) a = 1.0f / (1.0f + __expf(-a));    // sigmoid mask
      om[((size_t)(b * OMCH + m) << 14) + rem] = a;
    }
  }
}

// ============================================================================
// Stage 3: deformable conv as implicit GEMM with V_WMMA_F32_16X16X32_BF16.
// Block = 16 output pixels; 8 waves cover all 128 out-channels.
// K dim = 576 = 18 steps of 32.
// ============================================================================
__global__ __launch_bounds__(256) void dcn_wmma_kernel(
    const float* __restrict__ xin, const float* __restrict__ om,
    const __bf16* __restrict__ wb, const float* __restrict__ bias,
    float* __restrict__ out) {
  // im2col tile, bf16, pre-swizzled for WMMA B layout:
  // cm[(kidx>>4)*256 + pix*16 + (kidx&15)]  (kidx = c*9+k)
  __shared__ __align__(32) __bf16 cm[KC_ * 16];       // 18 KB
  __shared__ int   s_y0[K_ * 16], s_x0[K_ * 16];
  __shared__ float s_wy[K_ * 16], s_wx[K_ * 16], s_mk[K_ * 16];

  const int tid   = threadIdx.x;
  const int tile  = blockIdx.x;                       // 16 pixels per tile
  const int pbase = tile << 4;
  const int b     = pbase >> 14;
  const int rem0  = pbase & (HW_ - 1);

  // ---- phase 1: per-(tap,pixel) sampling metadata (144 entries) ----
  if (tid < K_ * 16) {
    int k = tid >> 4, pix = tid & 15;
    int rem = rem0 + pix;
    int y = rem >> 7, x = rem & (W_ - 1);
    size_t obase = (((size_t)b * OMCH) << 14) + rem;
    float dy = om[obase + ((size_t)(2 * k)     << 14)];
    float dx = om[obase + ((size_t)(2 * k + 1) << 14)];
    float mk = om[obase + ((size_t)(18 + k)    << 14)];
    float py = (float)(y + k / 3 - 1) + dy;           // ky + gy + off_y
    float px = (float)(x + k % 3 - 1) + dx;           // kx + gx + off_x
    float y0f = floorf(py), x0f = floorf(px);
    s_y0[tid] = (int)y0f; s_x0[tid] = (int)x0f;
    s_wy[tid] = py - y0f; s_wx[tid] = px - x0f;
    s_mk[tid] = mk;
  }
  __syncthreads();

  // ---- phase 2: modulated bilinear im2col -> LDS (576 x 16, bf16) ----
  for (int q = tid; q < KC_ * 16; q += 256) {
    int kidx = q >> 4;                                // c*9 + k
    int pix  = q & 15;
    int c = kidx / 9, k = kidx - c * 9;
    int mi = (k << 4) + pix;
    int y0 = s_y0[mi], x0 = s_x0[mi];
    float wy = s_wy[mi], wx = s_wx[mi];
    const float* src = xin + (((size_t)(b * C_ + c)) << 14);
    float v00 = 0.f, v01 = 0.f, v10 = 0.f, v11 = 0.f;
    if ((unsigned)y0 < (unsigned)H_) {
      if ((unsigned)x0       < (unsigned)W_) v00 = src[y0 * W_ + x0];
      if ((unsigned)(x0 + 1) < (unsigned)W_) v01 = src[y0 * W_ + x0 + 1];
    }
    if ((unsigned)(y0 + 1) < (unsigned)H_) {
      if ((unsigned)x0       < (unsigned)W_) v10 = src[(y0 + 1) * W_ + x0];
      if ((unsigned)(x0 + 1) < (unsigned)W_) v11 = src[(y0 + 1) * W_ + x0 + 1];
    }
    float v = (1.f - wy) * ((1.f - wx) * v00 + wx * v01)
            +        wy  * ((1.f - wx) * v10 + wx * v11);
    v *= s_mk[mi];
    cm[((kidx >> 4) << 8) + (pix << 4) + (kidx & 15)] = f2bf(v);
  }
  __syncthreads();

  // ---- phase 3: WMMA accumulation over K = 18 x 32 ----
  const int wave = tid >> 5;
  const int lane = tid & 31;
  const int mrow = (wave << 4) + (lane & 15);   // A row (out channel)
  const int koff = (lane >> 4) << 4;            // A layout: lanes<16 ->0, >=16 ->16
  const __bf16* wrow = wb + (size_t)mrow * KC_;
  __builtin_prefetch(wrow, 0, 0);               // global_prefetch of weight row

  f32x8_t acc = {};
#pragma unroll
  for (int kk = 0; kk < 18; ++kk) {
    // A fragment (16x32 bf16, contiguous k = 16*half + e, same as B)
    v8bf_t lo = *(const v8bf_t*)(wrow + kk * 32 + koff);
    v8bf_t hi = *(const v8bf_t*)(wrow + kk * 32 + 8 + koff);
    v16bf_t a;
#pragma unroll
    for (int e = 0; e < 8; ++e) { a[e] = lo[e]; a[e + 8] = hi[e]; }

    // B fragment from LDS (lane = pixel column; lane halves = K 16-chunks)
    const __bf16* bp = cm + ((((kk << 1) + (lane >> 4)) << 8) + ((lane & 15) << 4));
    v16bf_t bfrag = *(const v16bf_t*)bp;

    acc = __builtin_amdgcn_wmma_f32_16x16x32_bf16(
        /*neg_a=*/false, a, /*neg_b=*/false, bfrag,
        /*c_mod=*/(short)0, acc, /*reuse_a=*/false, /*reuse_b=*/false);
  }

  // ---- writeout: f32 C/D layout + bias ----
  const int n     = lane & 15;                  // pixel column
  const int mhalf = (lane >> 4) << 3;
  const int remn  = rem0 + n;
#pragma unroll
  for (int r = 0; r < 8; ++r) {
    int m = (wave << 4) + mhalf + r;
    out[(((size_t)(b * O_ + m)) << 14) + remn] = acc[r] + bias[m];
  }
}

// ============================================================================
extern "C" void kernel_launch(void* const* d_in, const int* in_sizes, int n_in,
                              void* d_out, int out_size, void* d_ws, size_t ws_size,
                              hipStream_t stream) {
  (void)in_sizes; (void)n_in; (void)out_size; (void)ws_size;
  const float* input_feat = (const float*)d_in[0];
  const float* inter      = (const float*)d_in[1];
  const float* weight     = (const float*)d_in[2];
  const float* bias       = (const float*)d_in[3];
  const float* w_om       = (const float*)d_in[4];
  const float* b_om       = (const float*)d_in[5];
  float* out = (float*)d_out;

  // workspace layout (all 16B aligned):
  //   om   : f32 [4][27][16384]            = 7,077,888 B
  //   wb   : bf16 [128][576]               =   147,456 B
  //   wompk: f32 [32][1152]                =   147,456 B
  char* ws = (char*)d_ws;
  float*  om    = (float*)ws;
  __bf16* wb    = (__bf16*)(ws + (size_t)B_ * OMCH * HW_ * sizeof(float));
  float*  wompk = (float*)(ws + (size_t)B_ * OMCH * HW_ * sizeof(float)
                              + (size_t)O_ * KC_ * sizeof(__bf16));

  pack_w_kernel<<<(O_ * KC_ + 255) / 256, 256, 0, stream>>>(
      weight, wb, O_ * KC_);
  pack_wom_kernel<<<(OMM * OMK + 255) / 256, 256, 0, stream>>>(
      w_om, wompk);
  om_wmma_kernel<<<(B_ * HW_) / 64, 256, 0, stream>>>(
      input_feat, inter, wompk, b_om, om);
  dcn_wmma_kernel<<<(B_ * HW_) / 16, 256, 0, stream>>>(
      input_feat, om, wb, bias, out);
}
